// EncoderTransformer_38431367365207
// MI455X (gfx1250) — matrix operand order repaired and checked by reading, not compiled, not run on hardware
//
#include <hip/hip_runtime.h>
#include <hip/hip_bf16.h>
#include <stdint.h>

// ---------------- problem constants (match reference) ----------------
#define S_   2048
#define IN_  64
#define D_   1024
#define OUT_ 128
#define H_   16
#define HD_  64
#define L_   4
#define FFN_ 4096
#define WIN_ 128
#define DOC_ 256

// ---------------- types for WMMA ----------------
typedef __attribute__((ext_vector_type(16))) __bf16 bf16x16;
typedef __attribute__((ext_vector_type(8)))  __bf16 bf16x8;
typedef __attribute__((ext_vector_type(8)))  float  f32x8;

__device__ __forceinline__ unsigned short f2bf(float f) {
    unsigned int u = __float_as_uint(f);
    unsigned int r = u + 0x7FFFu + ((u >> 16) & 1u);   // round-to-nearest-even
    return (unsigned short)(r >> 16);
}

// ---------------- CDNA5 async global->LDS copy (guarded) ----------------
#if __has_builtin(__builtin_amdgcn_global_load_async_to_lds_b128) && \
    __has_builtin(__builtin_amdgcn_s_wait_asynccnt)
#define USE_ASYNC_LDS 1
#else
#define USE_ASYNC_LDS 0
#endif

typedef int v4i_ __attribute__((vector_size(16)));
typedef __attribute__((address_space(1))) v4i_* g_v4i_ptr;
typedef __attribute__((address_space(3))) v4i_* l_v4i_ptr;

__device__ __forceinline__ void copy16_g2l(const unsigned short* g, unsigned short* l) {
#if USE_ASYNC_LDS
    __builtin_amdgcn_global_load_async_to_lds_b128(
        (g_v4i_ptr)(uintptr_t)g, (l_v4i_ptr)l, 0, 0);
#else
    __builtin_prefetch(g, 0, 1);
    *(uint4*)l = *(const uint4*)g;
#endif
}

__device__ __forceinline__ void copy_wait() {
#if USE_ASYNC_LDS
    __builtin_amdgcn_s_wait_asynccnt(0);
#endif
}

// ---------------- bf16 WMMA GEMM ----------------
// C[M,N] = A[M,K] @ B[K,N] (+bias)(+residual), A bf16 row-major MxK,
// Bt bf16 row-major NxK (pre-transposed). M%128==0, N%128==0, K%32==0.
#define BM 128
#define BN 128
#define BK 32

__global__ __launch_bounds__(256)
void gemm_bf16_wmma(const unsigned short* __restrict__ A,
                    const unsigned short* __restrict__ Bt,
                    const float* __restrict__ bias, const float* __restrict__ residual,
                    float* __restrict__ C, int M, int N, int K)
{
    __shared__ __align__(64) unsigned short lA[2][BM * BK];  // [m][k]
    __shared__ __align__(64) unsigned short lB[2][BN * BK];  // [n][k]

    const int tid   = threadIdx.x;
    const int lane  = tid & 31;
    const int wave  = tid >> 5;
    const int wm    = wave & 3;    // 4 waves along M
    const int wn    = wave >> 2;   // 2 waves along N
    const int rowBase = blockIdx.y * BM;
    const int colBase = blockIdx.x * BN;

    f32x8 acc[2][4];
#pragma unroll
    for (int i = 0; i < 2; ++i)
#pragma unroll
        for (int j = 0; j < 4; ++j) acc[i][j] = (f32x8){};

    const int halfl = lane & 15;
    const int aoff  = (lane >= 16) ? 16 : 0;   // A fragment contiguous 16-K run
    const int koff  = (lane >= 16) ? 16 : 0;   // B fragment contiguous 16-K run

    // per-thread tile-copy geometry: 512 16B chunks per 128x32 bf16 tile,
    // 256 threads x 2 chunks; chunk q -> row q>>2, 8-elem group q&3
    const int q0r = tid >> 2,         q0c = (tid & 3) * 8;
    const int q1r = (tid + 256) >> 2, q1c = ((tid + 256) & 3) * 8;

    auto load_tiles = [&](int buf, int k0) {
        copy16_g2l(A  + (size_t)(rowBase + q0r) * K + k0 + q0c, &lA[buf][q0r * BK + q0c]);
        copy16_g2l(Bt + (size_t)(colBase + q0r) * K + k0 + q0c, &lB[buf][q0r * BK + q0c]);
        copy16_g2l(A  + (size_t)(rowBase + q1r) * K + k0 + q1c, &lA[buf][q1r * BK + q1c]);
        copy16_g2l(Bt + (size_t)(colBase + q1r) * K + k0 + q1c, &lB[buf][q1r * BK + q1c]);
    };

    // preload first tile
    load_tiles(0, 0);
    copy_wait();
    __syncthreads();

    int cur = 0;
    for (int k0 = 0; k0 < K; k0 += BK) {
        if (k0 + BK < K) load_tiles(cur ^ 1, k0 + BK);  // overlap next DMA with WMMA

        bf16x16 afrag[2];
#pragma unroll
        for (int mt = 0; mt < 2; ++mt) {
            int r = wm * 32 + mt * 16 + halfl;
            bf16x8 a0 = *reinterpret_cast<const bf16x8*>(&lA[cur][r * BK + aoff]);
            bf16x8 a1 = *reinterpret_cast<const bf16x8*>(&lA[cur][r * BK + aoff + 8]);
            afrag[mt] = __builtin_shufflevector(a0, a1,
                0, 1, 2, 3, 4, 5, 6, 7, 8, 9, 10, 11, 12, 13, 14, 15);
        }
#pragma unroll
        for (int nt = 0; nt < 4; ++nt) {
            int c = wn * 64 + nt * 16 + halfl;
            bf16x16 bfrag = *reinterpret_cast<const bf16x16*>(&lB[cur][c * BK + koff]);
#pragma unroll
            for (int mt = 0; mt < 2; ++mt) {
                acc[mt][nt] = __builtin_amdgcn_wmma_f32_16x16x32_bf16(
                    false, afrag[mt], false, bfrag, (short)0, acc[mt][nt], false, false);
            }
        }

        copy_wait();        // this wave's async copies into (cur^1) done
        __syncthreads();    // everyone done reading cur & writing cur^1
        cur ^= 1;
    }

    // epilogue: C/D layout row = vgpr + 8*(lane>=16), col = lane%16
    const int rAdd = (lane >= 16) ? 8 : 0;
#pragma unroll
    for (int mt = 0; mt < 2; ++mt) {
#pragma unroll
        for (int nt = 0; nt < 4; ++nt) {
            int col = colBase + wn * 64 + nt * 16 + halfl;
            float bsv = bias ? bias[col] : 0.0f;
#pragma unroll
            for (int r = 0; r < 8; ++r) {
                int row = rowBase + wm * 32 + mt * 16 + r + rAdd;
                float vv = acc[mt][nt][r] + bsv;
                if (residual) vv += residual[(size_t)row * N + col];
                C[(size_t)row * N + col] = vv;
            }
        }
    }
}

// ---------------- fp32 -> bf16 convert ----------------
__global__ __launch_bounds__(256)
void cvt_bf16_kernel(const float* __restrict__ in, unsigned short* __restrict__ out, int n)
{
    int i = blockIdx.x * 256 + threadIdx.x;
    if (i < n) out[i] = f2bf(in[i]);
}

// ---------------- fp32 RxC -> bf16 CxR transpose (R,C multiples of 32) ----------------
__global__ __launch_bounds__(256)
void cvt_transpose_kernel(const float* __restrict__ in, unsigned short* __restrict__ out,
                          int R, int C)
{
    __shared__ float tile[32][33];
    int rb = blockIdx.y * 32, cb = blockIdx.x * 32;
    int tx = threadIdx.x & 31, ty = threadIdx.x >> 5;   // 32 x 8
#pragma unroll
    for (int i = 0; i < 4; ++i)
        tile[ty + i * 8][tx] = in[(size_t)(rb + ty + i * 8) * C + cb + tx];
    __syncthreads();
#pragma unroll
    for (int i = 0; i < 4; ++i)
        out[(size_t)(cb + ty + i * 8) * R + rb + tx] = f2bf(tile[tx][ty + i * 8]);
}

// ---------------- RMSNorm: fp32 in, bf16 out (one block per row) ----------------
__global__ __launch_bounds__(256)
void rmsnorm_kernel(const float* __restrict__ x, const float* __restrict__ w,
                    unsigned short* __restrict__ y, int D)
{
    int row = blockIdx.x;
    const float* xr = x + (size_t)row * D;
    float ss = 0.0f;
    for (int i = threadIdx.x; i < D; i += 256) { float v = xr[i]; ss += v * v; }
    __shared__ float red[256];
    red[threadIdx.x] = ss;
    __syncthreads();
    for (int s = 128; s > 0; s >>= 1) {
        if (threadIdx.x < s) red[threadIdx.x] += red[threadIdx.x + s];
        __syncthreads();
    }
    float scale = rsqrtf(red[0] / (float)D + 1e-5f);
    unsigned short* yr = y + (size_t)row * D;
    for (int i = threadIdx.x; i < D; i += 256) yr[i] = f2bf(xr[i] * scale * w[i]);
}

// ---------------- RoPE in place on fp32 (S,H,HD) buffer ----------------
__global__ __launch_bounds__(256)
void rope_kernel(float* __restrict__ buf, const int* __restrict__ tok_id)
{
    int idx = blockIdx.x * 256 + threadIdx.x;     // S*H*HD/2 threads
    const int half = HD_ / 2;
    int i = idx % half;
    int h = (idx / half) % H_;
    int s = idx / (half * H_);
    float t = (float)tok_id[s];
    float freq = __powf(10000.0f, -((float)(2 * i)) / (float)HD_);
    float ang = t * freq;
    float c = __cosf(ang), sn = __sinf(ang);
    size_t base = ((size_t)s * H_ + h) * HD_;
    float x1 = buf[base + i], x2 = buf[base + i + half];
    buf[base + i]        = x1 * c - x2 * sn;
    buf[base + i + half] = x1 * sn + x2 * c;
}

// ---------------- doc-windowed attention: one wave per (s,h), bf16 out ----------------
__global__ __launch_bounds__(256)
void attn_kernel(const float* __restrict__ q, const float* __restrict__ k,
                 const float* __restrict__ v, const int* __restrict__ doc_id,
                 const int* __restrict__ tok_id, unsigned short* __restrict__ o)
{
    __shared__ float sc[8][DOC_];
    const int lane = threadIdx.x & 31;
    const int wave = threadIdx.x >> 5;
    const int idx  = blockIdx.x * 8 + wave;
    const int s = idx / H_;
    const int h = idx % H_;
    const int doc = doc_id[s];
    const int tq  = tok_id[s];
    const int docStart = s - tq;               // 256-token doc block
    const float scale = 0.125f;                // 1/sqrt(64)
    const size_t qbase = ((size_t)s * H_ + h) * HD_;

    for (int t = 0; t < DOC_ / 32; ++t) {
        int j = docStart + lane + 32 * t;
        size_t kbase = ((size_t)j * H_ + h) * HD_;
        float dot = 0.0f;
#pragma unroll 8
        for (int d = 0; d < HD_; ++d) dot += q[qbase + d] * k[kbase + d];
        int dt = tok_id[j] - tq; if (dt < 0) dt = -dt;
        bool ok = (doc_id[j] == doc) && (dt < WIN_);
        sc[wave][lane + 32 * t] = ok ? dot * scale : -3.0e38f;
    }

    float m = -3.0e38f;
    for (int t = 0; t < DOC_ / 32; ++t) m = fmaxf(m, sc[wave][lane + 32 * t]);
    for (int i = 16; i > 0; i >>= 1) m = fmaxf(m, __shfl_xor(m, i, 32));
    float ssum = 0.0f;
    for (int t = 0; t < DOC_ / 32; ++t) {
        float p = __expf(sc[wave][lane + 32 * t] - m);
        sc[wave][lane + 32 * t] = p;
        ssum += p;
    }
    for (int i = 16; i > 0; i >>= 1) ssum += __shfl_xor(ssum, i, 32);
    float inv = 1.0f / ssum;
    __syncthreads();

    int d0 = lane * 2;
    float a0 = 0.0f, a1 = 0.0f;
    for (int j = 0; j < DOC_; ++j) {
        float p = sc[wave][j] * inv;
        size_t vbase = ((size_t)(docStart + j) * H_ + h) * HD_;
        a0 += p * v[vbase + d0];
        a1 += p * v[vbase + d0 + 1];
    }
    o[qbase + d0]     = f2bf(a0);
    o[qbase + d0 + 1] = f2bf(a1);
}

// ---------------- SiLU gate: out_bf16 = silu(a) * b ----------------
__global__ __launch_bounds__(256)
void silu_mul_kernel(const float* __restrict__ a, const float* __restrict__ b,
                     unsigned short* __restrict__ out, int n)
{
    int i = blockIdx.x * 256 + threadIdx.x;
    if (i < n) {
        float x = a[i];
        out[i] = f2bf((x / (1.0f + __expf(-x))) * b[i]);
    }
}

// ---------------- driver ----------------
extern "C" void kernel_launch(void* const* d_in, const int* in_sizes, int n_in,
                              void* d_out, int out_size, void* d_ws, size_t ws_size,
                              hipStream_t stream)
{
    const float* x           = (const float*)d_in[0];
    const float* emb_w       = (const float*)d_in[1];
    const float* emb_b       = (const float*)d_in[2];
    const float* wq          = (const float*)d_in[3];
    const float* wk          = (const float*)d_in[4];
    const float* wv          = (const float*)d_in[5];
    const float* wo          = (const float*)d_in[6];
    const float* attn_norm_w = (const float*)d_in[7];
    const float* ffn_norm_w  = (const float*)d_in[8];
    const float* w1          = (const float*)d_in[9];
    const float* w2          = (const float*)d_in[10];
    const float* w3          = (const float*)d_in[11];
    const float* out_norm_w  = (const float*)d_in[12];
    const float* out_w       = (const float*)d_in[13];
    const int*   doc_id      = (const int*)d_in[14];
    const int*   tok_id      = (const int*)d_in[15];

    const size_t SD = (size_t)S_ * D_;
    const size_t SF = (size_t)S_ * FFN_;
    const size_t DD = (size_t)D_ * D_;
    const size_t DF = (size_t)D_ * FFN_;

    // workspace carve-up
    char* cur = (char*)d_ws;
    auto carve = [&](size_t bytes) { char* p = cur; cur += (bytes + 255) & ~(size_t)255; return p; };
    float* h   = (float*)carve(SD * 4);
    float* qf  = (float*)carve(SD * 4);
    float* kf  = (float*)carve(SD * 4);
    float* vf  = (float*)carve(SD * 4);
    float* f1  = (float*)carve(SF * 4);
    float* f3  = (float*)carve(SF * 4);
    unsigned short* hnb  = (unsigned short*)carve(SD * 2);
    unsigned short* aob  = (unsigned short*)carve(SD * 2);
    unsigned short* f1b  = (unsigned short*)carve(SF * 2);
    unsigned short* xb   = (unsigned short*)carve((size_t)S_ * IN_ * 2);
    unsigned short* embT = (unsigned short*)carve((size_t)IN_ * D_ * 2);
    unsigned short* wqT  = (unsigned short*)carve((size_t)L_ * DD * 2);
    unsigned short* wkT  = (unsigned short*)carve((size_t)L_ * DD * 2);
    unsigned short* wvT  = (unsigned short*)carve((size_t)L_ * DD * 2);
    unsigned short* woT  = (unsigned short*)carve((size_t)L_ * DD * 2);
    unsigned short* w1T  = (unsigned short*)carve((size_t)L_ * DF * 2);
    unsigned short* w3T  = (unsigned short*)carve((size_t)L_ * DF * 2);
    unsigned short* w2T  = (unsigned short*)carve((size_t)L_ * DF * 2);
    unsigned short* outT = (unsigned short*)carve((size_t)D_ * OUT_ * 2);

    dim3 blk(256);
    auto transpose = [&](const float* in, unsigned short* out, int R, int C) {
        dim3 grid(C / 32, R / 32);
        cvt_transpose_kernel<<<grid, blk, 0, stream>>>(in, out, R, C);
    };
    auto gemm = [&](const unsigned short* A, const unsigned short* Bt, const float* bias,
                    const float* res, float* C, int M, int N, int K) {
        dim3 grid(N / BN, M / BM);
        gemm_bf16_wmma<<<grid, blk, 0, stream>>>(A, Bt, bias, res, C, M, N, K);
    };

    // one-time (per launch) operand preparation: bf16 + B-transpose
    cvt_bf16_kernel<<<(S_ * IN_) / 256, blk, 0, stream>>>(x, xb, S_ * IN_);
    transpose(emb_w, embT, IN_, D_);
    transpose(out_w, outT, D_, OUT_);
    for (int l = 0; l < L_; ++l) {
        transpose(wq + l * DD, wqT + l * DD, D_, D_);
        transpose(wk + l * DD, wkT + l * DD, D_, D_);
        transpose(wv + l * DD, wvT + l * DD, D_, D_);
        transpose(wo + l * DD, woT + l * DD, D_, D_);
        transpose(w1 + l * DF, w1T + l * DF, D_, FFN_);
        transpose(w3 + l * DF, w3T + l * DF, D_, FFN_);
        transpose(w2 + l * DF, w2T + l * DF, FFN_, D_);
    }

    // embedding: h = x @ emb_w + emb_b
    gemm(xb, embT, emb_b, nullptr, h, S_, D_, IN_);

    const int ropeBlocks = (S_ * H_ * (HD_ / 2)) / 256;
    for (int l = 0; l < L_; ++l) {
        rmsnorm_kernel<<<S_, blk, 0, stream>>>(h, attn_norm_w + (size_t)l * D_, hnb, D_);
        gemm(hnb, wqT + l * DD, nullptr, nullptr, qf, S_, D_, D_);
        gemm(hnb, wkT + l * DD, nullptr, nullptr, kf, S_, D_, D_);
        gemm(hnb, wvT + l * DD, nullptr, nullptr, vf, S_, D_, D_);
        rope_kernel<<<ropeBlocks, blk, 0, stream>>>(qf, tok_id);
        rope_kernel<<<ropeBlocks, blk, 0, stream>>>(kf, tok_id);
        attn_kernel<<<(S_ * H_) / 8, blk, 0, stream>>>(qf, kf, vf, doc_id, tok_id, aob);
        gemm(aob, woT + l * DD, nullptr, h, h, S_, D_, D_);          // fused residual

        rmsnorm_kernel<<<S_, blk, 0, stream>>>(h, ffn_norm_w + (size_t)l * D_, hnb, D_);
        gemm(hnb, w1T + l * DF, nullptr, nullptr, f1, S_, FFN_, D_);
        gemm(hnb, w3T + l * DF, nullptr, nullptr, f3, S_, FFN_, D_);
        silu_mul_kernel<<<(int)(SF / 256), blk, 0, stream>>>(f1, f3, f1b, (int)SF);
        gemm(f1b, w2T + l * DF, nullptr, h, h, S_, D_, FFN_);        // fused residual
    }

    rmsnorm_kernel<<<S_, blk, 0, stream>>>(h, out_norm_w, hnb, D_);
    gemm(hnb, outT, nullptr, nullptr, (float*)d_out, S_, OUT_, D_);
}
